// Model_39676907886731
// MI455X (gfx1250) — matrix operand order repaired and checked by reading, not compiled, not run on hardware
//
#include <hip/hip_runtime.h>
#include <hip/hip_bf16.h>

// ---------------------------------------------------------------------------
// MI455X (gfx1250) attention:  Q=xWq^T, K=xWk^T, V=xWv^T (fp32 -> bf16),
// S = (Q K^T) * sqrt(F)  (f32, 134MB -> lives in 192MB L2),
// P = softmax(S) packed bf16 in place, O = P V  via Vt[b][g][j].
// All GEMMs through v_wmma_f32_16x16x32_bf16, wave32, 128x128x64 tiles.
// ---------------------------------------------------------------------------

typedef __bf16 bf16;
typedef __attribute__((ext_vector_type(16))) __bf16 v16bf;
typedef __attribute__((ext_vector_type(4)))  __bf16 v4bf;
typedef __attribute__((ext_vector_type(8)))  float  v8f;

union FragAB { v16bf v; v4bf q[4]; };
union AccU   { v8f v; float f[8]; };

#define BM 128
#define BN 128
#define BK 64
#define LDP 72   // LDS pitch (bf16): 144B row stride = 36 banks -> conflict-free frag loads, 8B aligned

#define OUT_BF16        0
#define OUT_F32         1
#define OUT_TRANS_BF16  2   // V projection: write only Vt[b][g][j] (b=row>>11, j=row&2047)

template<typename TA, int OUTMODE>
__global__ __launch_bounds__(256) void gemm_nt_wmma(
    const TA* __restrict__ A, const TA* __restrict__ Bm,
    void* __restrict__ Cv, bf16* __restrict__ Ct,
    int lda, int ldb, int ldc,
    long long sA, long long sB, long long sC,
    int K, float alpha)
{
    __shared__ bf16 Alds[BM * LDP];
    __shared__ bf16 Blds[BN * LDP];

    const int t     = threadIdx.x;
    const int lane  = t & 31;
    const int wid   = t >> 5;
    const int waveM = wid & 3;    // 4 waves along M
    const int waveN = wid >> 2;   // 2 waves along N
    const int h     = lane >> 4;  // K-half selector
    const int rr    = lane & 15;  // row (A) / col (B) within 16

    const long long z = blockIdx.z;
    const TA* Ab = A  + z * sA;
    const TA* Bb = Bm + z * sB;

    const int ldRow = t >> 1;           // 0..127: LDS row this thread fills
    const int kc    = (t & 1) * 32;     // 32 K-elems per thread per tile

    v8f acc[2][4];
    const v8f vzero = {0.f,0.f,0.f,0.f,0.f,0.f,0.f,0.f};
    #pragma unroll
    for (int i = 0; i < 2; ++i)
        #pragma unroll
        for (int j = 0; j < 4; ++j) acc[i][j] = vzero;

    const int gRowA = blockIdx.x * BM + ldRow;
    const int gRowB = blockIdx.y * BN + ldRow;

    for (int k0 = 0; k0 < K; k0 += BK) {
        __syncthreads();
        {   // cooperative global -> LDS (fp32 inputs converted to bf16 on the fly)
            const TA* srcA = Ab + (size_t)gRowA * lda + k0 + kc;
            const TA* srcB = Bb + (size_t)gRowB * ldb + k0 + kc;
            bf16* dA = &Alds[ldRow * LDP + kc];
            bf16* dB = &Blds[ldRow * LDP + kc];
            if constexpr (sizeof(TA) == 4) {
                const float4* a4 = (const float4*)srcA;
                const float4* b4 = (const float4*)srcB;
                #pragma unroll
                for (int c = 0; c < 8; ++c) {
                    float4 fa = a4[c], fb = b4[c];
                    v4bf pa = {(bf16)fa.x, (bf16)fa.y, (bf16)fa.z, (bf16)fa.w};
                    v4bf pb = {(bf16)fb.x, (bf16)fb.y, (bf16)fb.z, (bf16)fb.w};
                    *(v4bf*)(dA + 4 * c) = pa;
                    *(v4bf*)(dB + 4 * c) = pb;
                }
            } else {
                #pragma unroll
                for (int c = 0; c < 8; ++c) {
                    *(v4bf*)(dA + 4 * c) = *(const v4bf*)(srcA + 4 * c);
                    *(v4bf*)(dB + 4 * c) = *(const v4bf*)(srcB + 4 * c);
                }
            }
            if (k0 + BK < K) {   // gfx1250 global_prefetch_b8 for next K tile
                __builtin_prefetch(srcA + BK, 0, 0);
                __builtin_prefetch(srcB + BK, 0, 0);
            }
        }
        __syncthreads();

        #pragma unroll
        for (int kk = 0; kk < BK; kk += 32) {
            FragAB af[2], bq[4];
            // A 16x32 bf16 layout: lanes 0-15 hold K 0-7 & 16-23, lanes 16-31 K 8-15 & 24-31
            #pragma unroll
            for (int i = 0; i < 2; ++i) {
                const bf16* p = &Alds[(waveM * 32 + i * 16 + rr) * LDP + kk + h * 8];
                af[i].q[0] = *(const v4bf*)(p);
                af[i].q[1] = *(const v4bf*)(p + 4);
                af[i].q[2] = *(const v4bf*)(p + 16);
                af[i].q[3] = *(const v4bf*)(p + 20);
            }
            // B 32x16 bf16 layout: lanes 0-15 hold K 0-7 & 16-23, lanes 16-31 K 8-15 & 24-31 (col = lane&15)
            #pragma unroll
            for (int j = 0; j < 4; ++j) {
                const bf16* p = &Blds[(waveN * 64 + j * 16 + rr) * LDP + kk + h * 8];
                bq[j].q[0] = *(const v4bf*)(p);
                bq[j].q[1] = *(const v4bf*)(p + 4);
                bq[j].q[2] = *(const v4bf*)(p + 16);
                bq[j].q[3] = *(const v4bf*)(p + 20);
            }
            #pragma unroll
            for (int i = 0; i < 2; ++i)
                #pragma unroll
                for (int j = 0; j < 4; ++j)
                    acc[i][j] = __builtin_amdgcn_wmma_f32_16x16x32_bf16(
                        false, af[i].v, false, bq[j].v, (short)0, acc[i][j], false, false);
        }
    }

    // C/D layout: N = lane&15, M = (lane>>4)*8 + vgpr
    const int cRow0 = blockIdx.x * BM + waveM * 32 + h * 8;
    const int cCol0 = blockIdx.y * BN + waveN * 64 + rr;
    #pragma unroll
    for (int i = 0; i < 2; ++i) {
        #pragma unroll
        for (int j = 0; j < 4; ++j) {
            AccU a; a.v = acc[i][j];
            #pragma unroll
            for (int r = 0; r < 8; ++r) {
                const int row = cRow0 + i * 16 + r;
                const int col = cCol0 + j * 16;
                const float val = a.f[r] * alpha;
                if constexpr (OUTMODE == OUT_F32) {
                    ((float*)Cv)[(size_t)(z * sC) + (size_t)row * ldc + col] = val;
                } else if constexpr (OUTMODE == OUT_BF16) {
                    ((bf16*)Cv)[(size_t)row * ldc + col] = (bf16)val;
                } else { // OUT_TRANS_BF16: Vt[b][g=col][j=row&2047]
                    const int b  = row >> 11;
                    const int jj = row & 2047;
                    Ct[((size_t)(b * 1024 + col)) * 2048 + jj] = (bf16)val;
                }
            }
        }
    }
}

// One 256-thread block per row of S (2048 f32). Packs normalized P as bf16
// into the low half of the same row slot (row pitch stays 4096 bf16 elems).
__global__ __launch_bounds__(256) void softmax_rows(float* __restrict__ S)
{
    const size_t row = blockIdx.x;
    float* rp = S + row * 2048;
    const int t = threadIdx.x;

    float4 v0 = ((const float4*)rp)[t];
    float4 v1 = ((const float4*)rp)[t + 256];
    float x[8] = {v0.x, v0.y, v0.z, v0.w, v1.x, v1.y, v1.z, v1.w};

    float m = x[0];
    #pragma unroll
    for (int i = 1; i < 8; ++i) m = fmaxf(m, x[i]);
    #pragma unroll
    for (int o = 16; o > 0; o >>= 1) m = fmaxf(m, __shfl_xor(m, o, 32));

    __shared__ float red[8];
    if ((t & 31) == 0) red[t >> 5] = m;
    __syncthreads();
    float rm = red[0];
    #pragma unroll
    for (int i = 1; i < 8; ++i) rm = fmaxf(rm, red[i]);
    __syncthreads();

    float e[8], s = 0.f;
    #pragma unroll
    for (int i = 0; i < 8; ++i) { e[i] = __expf(x[i] - rm); s += e[i]; }
    #pragma unroll
    for (int o = 16; o > 0; o >>= 1) s += __shfl_xor(s, o, 32);
    if ((t & 31) == 0) red[t >> 5] = s;
    __syncthreads();
    float rs = 0.f;
    #pragma unroll
    for (int i = 0; i < 8; ++i) rs += red[i];
    const float inv = 1.0f / rs;

    bf16* bp = (bf16*)rp;   // in-place pack (all reads completed before barriers above)
    v4bf p0 = {(bf16)(e[0]*inv), (bf16)(e[1]*inv), (bf16)(e[2]*inv), (bf16)(e[3]*inv)};
    v4bf p1 = {(bf16)(e[4]*inv), (bf16)(e[5]*inv), (bf16)(e[6]*inv), (bf16)(e[7]*inv)};
    *(v4bf*)(bp + 4 * t)        = p0;
    *(v4bf*)(bp + 1024 + 4 * t) = p1;
}

extern "C" void kernel_launch(void* const* d_in, const int* in_sizes, int n_in,
                              void* d_out, int out_size, void* d_ws, size_t ws_size,
                              hipStream_t stream)
{
    const float* query = (const float*)d_in[0];
    const float* key   = (const float*)d_in[1];
    const float* value = (const float*)d_in[2];
    const float* Wq    = (const float*)d_in[3];
    const float* Wk    = (const float*)d_in[4];
    const float* Wv    = (const float*)d_in[5];

    const int B = 8, L = 2048, F = 1024;
    const size_t PROJ = (size_t)B * L * F;         // 16,777,216 elems

    // workspace: Q(bf16) | K(bf16) | Vt(bf16, [b][g][j]) | S(f32, P packed in place)
    bf16*  Qb = (bf16*)d_ws;
    bf16*  Kb = Qb + PROJ;
    bf16*  Vt = Kb + PROJ;
    float* S  = (float*)(Vt + PROJ);               // 8*2048*2048 f32 = 134MB (fits global L2)

    dim3 blk(256);

    // Projections: x @ W^T  (M=16384, N=1024, K=1024)
    gemm_nt_wmma<float, OUT_BF16><<<dim3(128, 8, 1), blk, 0, stream>>>(
        query, Wq, Qb, nullptr, F, F, F, 0, 0, 0, F, 1.0f);
    gemm_nt_wmma<float, OUT_BF16><<<dim3(128, 8, 1), blk, 0, stream>>>(
        key, Wk, Kb, nullptr, F, F, F, 0, 0, 0, F, 1.0f);
    gemm_nt_wmma<float, OUT_TRANS_BF16><<<dim3(128, 8, 1), blk, 0, stream>>>(
        value, Wv, nullptr, Vt, F, F, 0, 0, 0, 0, F, 1.0f);

    // S = (Q K^T) * sqrt(F)   (per batch: M=N=2048, K=1024; qk.div(F^-0.5) == *32)
    gemm_nt_wmma<bf16, OUT_F32><<<dim3(16, 16, 8), blk, 0, stream>>>(
        Qb, Kb, S, nullptr, F, F, L,
        (long long)L * F, (long long)L * F, (long long)L * L, F, 32.0f);

    // row softmax, pack P bf16 in place
    softmax_rows<<<dim3(B * L), blk, 0, stream>>>(S);

    // O = P V   (per batch: M=2048, N=1024, K=2048; A pitch 4096 bf16, B = Vt rows)
    gemm_nt_wmma<bf16, OUT_F32><<<dim3(16, 8, 8), blk, 0, stream>>>(
        (const bf16*)S, Vt, d_out, nullptr, 2 * L, L, F,
        (long long)L * 2 * L, (long long)F * L, (long long)L * F, L, 1.0f);
}
